// W8A8B8O8Linear_40243843564267
// MI455X (gfx1250) — 8-bit matrix operand order repaired and checked by reading, compiled, not run on hardware
//
#include <hip/hip_runtime.h>
#include <math.h>

typedef int v8i __attribute__((ext_vector_type(8)));
typedef int v4i __attribute__((ext_vector_type(4)));
typedef int v2i __attribute__((ext_vector_type(2)));

#define KDIM 1024
#define NDIM 1024
#define MDIM (4 * 8192)
#define MTILE 128
#define NCHUNK 128
#define SXS 1040
#define SWS 1040
#define XS_BYTES (MTILE * SXS)
#define WS_BYTES (NCHUNK * SWS)
#define LDS_BYTES (XS_BYTES + WS_BYTES)

__device__ __forceinline__ unsigned pack4(int a, int b, int c, int d) {
  return (unsigned)(a & 0xFF) | ((unsigned)(b & 0xFF) << 8) |
         ((unsigned)(c & 0xFF) << 16) | ((unsigned)d << 24);
}

__global__ __launch_bounds__(256) void w8a8_o8_gemm_kernel(
    const int* __restrict__ xg, const int* __restrict__ wg,
    const float* __restrict__ ag, const float* __restrict__ bg,
    float* __restrict__ out) {
  extern __shared__ unsigned char smem[];
  unsigned char* xs  = smem;
  unsigned char* wsm = smem + XS_BYTES;

  const int t    = threadIdx.x;
  const int m0   = blockIdx.x * MTILE;
  const int wave = t >> 5;
  const int lane = t & 31;
  const int half = lane >> 4;
  const int l15  = lane & 15;
  const int wm   = (wave >> 1) << 5;
  const int wn   = (wave & 1) << 6;

  const float alpha = ag[0] * bg[0];

  #pragma unroll 4
  for (int row = 0; row < MTILE; ++row) {
    v4i v = *reinterpret_cast<const v4i*>(xg + (size_t)(m0 + row) * KDIM + t * 4);
    *reinterpret_cast<unsigned*>(xs + row * SXS + t * 4) = pack4(v[0], v[1], v[2], v[3]);
  }

  for (int chunk = 0; chunk < NDIM / NCHUNK; ++chunk) {
    const int n0 = chunk * NCHUNK;
    if (chunk) __syncthreads();

    {
      const int c = t & 127;
      #pragma unroll 4
      for (int kk = 0; kk < KDIM / 4; kk += 2) {
        const int k4 = kk + (t >> 7);
        const int* src = wg + (size_t)(k4 * 4) * NDIM + n0 + c;
        *reinterpret_cast<unsigned*>(wsm + c * SWS + k4 * 4) =
            pack4(src[0], src[NDIM], src[2 * NDIM], src[3 * NDIM]);
      }
    }
    __syncthreads();

    v8i acc[2][4];
    #pragma unroll
    for (int r = 0; r < 2; ++r)
      #pragma unroll
      for (int c = 0; c < 4; ++c) {
        v8i z = {0, 0, 0, 0, 0, 0, 0, 0};
        acc[r][c] = z;
      }

    #pragma unroll 2
    for (int k0 = 0; k0 < KDIM; k0 += 64) {
      v8i Af[2], Bf[4];
      #pragma unroll
      for (int r = 0; r < 2; ++r) {
        const unsigned char* ab = xs + (wm + r * 16 + l15) * SXS + k0 + half * 8;
        v2i d0 = *reinterpret_cast<const v2i*>(ab);
        v2i d1 = *reinterpret_cast<const v2i*>(ab + 16);
        v2i d2 = *reinterpret_cast<const v2i*>(ab + 32);
        v2i d3 = *reinterpret_cast<const v2i*>(ab + 48);
        v8i a8 = {d0[0], d0[1], d1[0], d1[1], d2[0], d2[1], d3[0], d3[1]};
        Af[r] = a8;
      }
      #pragma unroll
      for (int c = 0; c < 4; ++c) {
        const unsigned char* bb = wsm + (wn + c * 16 + l15) * SWS + k0 + half * 8;
        v2i e0 = *reinterpret_cast<const v2i*>(bb);
        v2i e1 = *reinterpret_cast<const v2i*>(bb + 16);
        v2i e2 = *reinterpret_cast<const v2i*>(bb + 32);
        v2i e3 = *reinterpret_cast<const v2i*>(bb + 48);
        v8i b8 = {e0[0], e0[1], e1[0], e1[1], e2[0], e2[1], e3[0], e3[1]};
        Bf[c] = b8;
      }
      #pragma unroll
      for (int r = 0; r < 2; ++r)
        #pragma unroll
        for (int c = 0; c < 4; ++c)
          acc[r][c] = __builtin_amdgcn_wmma_i32_16x16x64_iu8(
               true, Af[r],  true, Bf[c],
              acc[r][c],  false,  false);
    }

    #pragma unroll
    for (int r = 0; r < 2; ++r) {
      #pragma unroll
      for (int c = 0; c < 4; ++c) {
        const int gr = m0 + wm + r * 16 + half * 8;
        const int gc = n0 + wn + c * 16 + l15;
        #pragma unroll
        for (int v = 0; v < 8; ++v) {
          float f = rintf((float)acc[r][c][v] * alpha);
          f = fminf(fmaxf(f, -128.0f), 127.0f);
          out[(size_t)(gr + v) * NDIM + gc] = f;
        }
      }
    }
  }
}

extern "C" void kernel_launch(void* const* d_in, const int* in_sizes, int n_in,
                              void* d_out, int out_size, void* d_ws, size_t ws_size,
                              hipStream_t stream) {
  const int*   x = (const int*)d_in[0];
  const int*   W = (const int*)d_in[1];
  const float* a = (const float*)d_in[2];
  const float* b = (const float*)d_in[3];
  float* out = (float*)d_out;
  (void)in_sizes; (void)n_in; (void)out_size; (void)d_ws; (void)ws_size;

  dim3 grid(MDIM / MTILE);
  dim3 block(256);
  hipLaunchKernelGGL(w8a8_o8_gemm_kernel, grid, block, LDS_BYTES, stream,
                     x, W, a, b, out);
}
